// ResidualTransformerRegressionModule_40312563040403
// MI455X (gfx1250) — matrix operand order repaired and checked by reading, not compiled, not run on hardware
//
#include <hip/hip_runtime.h>

// ---------------- constants ----------------
constexpr int BNUM = 8;     // clouds
constexpr int PNUM = 512;   // points per cloud
constexpr int KNN  = 20;    // neighbors
constexpr int DIM  = 512;   // hidden dim
constexpr int LAY  = 3;     // layers
constexpr int NTOT = BNUM * PNUM; // 4096

typedef __attribute__((ext_vector_type(16))) __bf16 v16bf;
typedef __attribute__((ext_vector_type(8)))  __bf16 v8bf;
typedef __attribute__((ext_vector_type(8)))  float  v8f;

// f32 -> bf16 (round to nearest even), stored as ushort
__device__ __forceinline__ unsigned short f2bf(float f) {
    unsigned int u = __float_as_uint(f);
    unsigned int r = u + 0x7FFFu + ((u >> 16) & 1u);
    return (unsigned short)(r >> 16);
}

// ---------------- KNN ----------------
// one block per cloud, 512 threads; positions staged in LDS; top-20 kept
// in registers via fully unrolled bubble insert (no scratch spills).
__global__ __launch_bounds__(512) void knn_kernel(const float* __restrict__ pos,
                                                  int* __restrict__ nbr) {
    int b = blockIdx.x;
    int p = threadIdx.x;
    __shared__ float sx[PNUM], sy[PNUM], sz[PNUM];
    int gi = b * PNUM + p;
    sx[p] = pos[gi * 3 + 0];
    sy[p] = pos[gi * 3 + 1];
    sz[p] = pos[gi * 3 + 2];
    __syncthreads();
    float qx = sx[p], qy = sy[p], qz = sz[p];
    float best[KNN];
    int   bidx[KNN];
#pragma unroll
    for (int t = 0; t < KNN; ++t) { best[t] = 3.0e38f; bidx[t] = 0; }
    for (int j = 0; j < PNUM; ++j) {
        float dx = qx - sx[j], dy = qy - sy[j], dz = qz - sz[j];
        float d2 = dx * dx + dy * dy + dz * dz;
        if (j == p) d2 = 3.0e38f;   // exclude self
        float cd = d2; int ci = j;
#pragma unroll
        for (int t = 0; t < KNN; ++t) {
            if (cd < best[t]) {
                float tf = best[t]; best[t] = cd; cd = tf;
                int   ti = bidx[t]; bidx[t] = ci; ci = ti;
            }
        }
    }
#pragma unroll
    for (int t = 0; t < KNN; ++t) nbr[gi * KNN + t] = b * PNUM + bidx[t];
}

// ---------------- small elementwise kernels ----------------
__global__ void ffm_kernel(const float* __restrict__ pos,
                           const float* __restrict__ W, const float* __restrict__ b,
                           float* __restrict__ x, unsigned short* __restrict__ xb) {
    int t = blockIdx.x * 256 + threadIdx.x;
    int i = t >> 9, d = t & 511;
    float v = pos[i * 3 + 0] * W[d] + pos[i * 3 + 1] * W[512 + d] +
              pos[i * 3 + 2] * W[1024 + d] + b[d];
    x[t] = v;
    xb[t] = f2bf(v);
}

__global__ void convert_bf16_kernel(const float* __restrict__ in,
                                    unsigned short* __restrict__ out, int n) {
    int t = blockIdx.x * 256 + threadIdx.x;
    if (t < n) out[t] = f2bf(in[t]);
}

// W [512,512] f32 row-major -> out bf16 [512,512] with out[c*512+r] = W[r*512+c]
__global__ void transpose_bf16_kernel(const float* __restrict__ W,
                                      unsigned short* __restrict__ out) {
    int t = blockIdx.x * 256 + threadIdx.x;   // t = c*512 + r
    int r = t & 511, c = t >> 9;
    out[t] = f2bf(W[r * 512 + c]);
}

// Wpa[r][n] = sum_d Wpos[r][d]*Wattn[d][n]  (r<3);  r==3 -> bpa[n]=sum_d bpos[d]*Wattn[d][n]+battn[n]
__global__ void wpa_kernel(const float* __restrict__ Wpos, const float* __restrict__ bpos,
                           const float* __restrict__ Wattn, const float* __restrict__ battn,
                           float* __restrict__ Wpa, float* __restrict__ bpa) {
    int t = blockIdx.x * 256 + threadIdx.x;   // 0..2047
    int r = t >> 9, n = t & 511;
    float acc = 0.0f;
    if (r < 3) {
        for (int d = 0; d < 512; ++d) acc += Wpos[r * 512 + d] * Wattn[d * 512 + n];
        Wpa[r * 512 + n] = acc;
    } else {
        for (int d = 0; d < 512; ++d) acc += bpos[d] * Wattn[d * 512 + n];
        bpa[n] = acc + battn[n];
    }
}

// ---------------- WMMA GEMM ----------------
// C[M,512] = A[M,512](bf16 rowmajor) @ W  where Bt[n][k] = W[k][n] (bf16)
// EPI 0: Cf = acc + bias
// EPI 1: w = resIn + resCoef[0]*(acc+bias); Cf = w; Cb = bf16(w)
// EPI 2: CbT[n*M+m] = bf16(acc)            (transposed bf16 output, no bias)
template <int EPI>
__global__ __launch_bounds__(256) void gemm_wmma(
    const unsigned short* __restrict__ A,
    const unsigned short* __restrict__ Bt,
    const float* __restrict__ bias,
    float* __restrict__ Cf,
    unsigned short* __restrict__ CbT,
    const float* __restrict__ resIn,
    const float* __restrict__ resCoef,
    unsigned short* __restrict__ Cb,
    int M) {
    const int Kd = 512;
    int tid  = threadIdx.x;
    int wave = tid >> 5, lane = tid & 31;
    int half = lane >> 4, lr = lane & 15;
    int wm = wave & 3, wn = wave >> 2;
    int Mb = blockIdx.x * 128, Nb = blockIdx.y * 64;

    const unsigned short* a0p = A  + (size_t)(Mb + wm * 32 + lr)      * Kd + half * 16;
    const unsigned short* a1p = A  + (size_t)(Mb + wm * 32 + 16 + lr) * Kd + half * 16;
    const unsigned short* b0p = Bt + (size_t)(Nb + wn * 32 + lr)      * Kd + half * 16;
    const unsigned short* b1p = Bt + (size_t)(Nb + wn * 32 + 16 + lr) * Kd + half * 16;

    v8f acc[2][2] = {};

    for (int k = 0; k < Kd; k += 32) {
        union U { v16bf v; v8bf h[2]; };
        U a0, a1, b0, b1;
        a0.h[0] = *(const v8bf*)(a0p + k);
        a0.h[1] = *(const v8bf*)(a0p + k + 8);
        a1.h[0] = *(const v8bf*)(a1p + k);
        a1.h[1] = *(const v8bf*)(a1p + k + 8);
        b0.h[0] = *(const v8bf*)(b0p + k);
        b0.h[1] = *(const v8bf*)(b0p + k + 8);
        b1.h[0] = *(const v8bf*)(b1p + k);
        b1.h[1] = *(const v8bf*)(b1p + k + 8);
        __builtin_prefetch(a0p + k + 64, 0, 1);
        __builtin_prefetch(a1p + k + 64, 0, 1);
        acc[0][0] = __builtin_amdgcn_wmma_f32_16x16x32_bf16(false, a0.v, false, b0.v,
                                                            (short)0, acc[0][0], false, false);
        acc[0][1] = __builtin_amdgcn_wmma_f32_16x16x32_bf16(false, a0.v, false, b1.v,
                                                            (short)0, acc[0][1], false, false);
        acc[1][0] = __builtin_amdgcn_wmma_f32_16x16x32_bf16(false, a1.v, false, b0.v,
                                                            (short)0, acc[1][0], false, false);
        acc[1][1] = __builtin_amdgcn_wmma_f32_16x16x32_bf16(false, a1.v, false, b1.v,
                                                            (short)0, acc[1][1], false, false);
    }

#pragma unroll
    for (int t = 0; t < 2; ++t) {
#pragma unroll
        for (int s = 0; s < 2; ++s) {
            int mrb = Mb + wm * 32 + t * 16 + half * 8;
            int n   = Nb + wn * 32 + s * 16 + lr;
            float bi = (EPI == 2) ? 0.0f : (bias ? bias[n] : 0.0f);
#pragma unroll
            for (int g = 0; g < 8; ++g) {
                int m = mrb + g;
                float val = acc[t][s][g] + bi;
                if constexpr (EPI == 0) {
                    Cf[(size_t)m * 512 + n] = val;
                } else if constexpr (EPI == 1) {
                    size_t idx = (size_t)m * 512 + n;
                    float w = resIn[idx] + resCoef[0] * val;
                    Cf[idx] = w;
                    Cb[idx] = f2bf(w);
                } else {
                    CbT[(size_t)n * M + m] = f2bf(val);
                }
            }
        }
    }
}

// ---------------- attention / aggregation ----------------
// alpha_ikd = u[i,d] - v[j,d] + rel_ik . Wpa[:,d] + bpa[d]
// softmax over k per (i,d); conv = sum_k alpha * (xl[j,d] + rel_ik . Wpos[:,d] + bpos[d])
// g = bf16(elu(conv))
__global__ __launch_bounds__(256) void attn_kernel(
    const float* __restrict__ pos, const int* __restrict__ nbr,
    const float* __restrict__ u, const float* __restrict__ v,
    const float* __restrict__ xl,
    const float* __restrict__ Wpa, const float* __restrict__ bpa,
    const float* __restrict__ Wpos, const float* __restrict__ bpos,
    unsigned short* __restrict__ g) {
    int i = blockIdx.x;
    int d = blockIdx.y * 256 + threadIdx.x;
    __shared__ int   sj[KNN];
    __shared__ float srx[KNN], sry[KNN], srz[KNN];
    if (threadIdx.x < KNN) {
        int j = nbr[i * KNN + threadIdx.x];
        sj[threadIdx.x]  = j;
        srx[threadIdx.x] = pos[i * 3 + 0] - pos[j * 3 + 0];
        sry[threadIdx.x] = pos[i * 3 + 1] - pos[j * 3 + 1];
        srz[threadIdx.x] = pos[i * 3 + 2] - pos[j * 3 + 2];
    }
    __syncthreads();

    float uid = u[(size_t)i * 512 + d];
    float w0 = Wpa[d], w1 = Wpa[512 + d], w2 = Wpa[1024 + d], bb = bpa[d];
    float a[KNN];
    float mx = -3.0e38f;
#pragma unroll
    for (int k = 0; k < KNN; ++k) {
        int j = sj[k];
        float t = uid - v[(size_t)j * 512 + d] +
                  srx[k] * w0 + sry[k] * w1 + srz[k] * w2 + bb;
        a[k] = t;
        mx = fmaxf(mx, t);
    }
    float p0 = Wpos[d], p1 = Wpos[512 + d], p2 = Wpos[1024 + d], pb = bpos[d];
    float s = 0.0f, conv = 0.0f;
#pragma unroll
    for (int k = 0; k < KNN; ++k) {
        float e = __expf(a[k] - mx);
        s += e;
        int j = sj[k];
        float delta = srx[k] * p0 + sry[k] * p1 + srz[k] * p2 + pb;
        conv += e * (xl[(size_t)j * 512 + d] + delta);
    }
    conv /= s;
    float gg = conv > 0.0f ? conv : (__expf(conv) - 1.0f);   // ELU
    g[(size_t)i * 512 + d] = f2bf(gg);
}

// ---------------- head ----------------
__global__ void pool_max_kernel(const float* __restrict__ h1, float* __restrict__ pooled) {
    int b = blockIdx.x;
    int d = blockIdx.y * 256 + threadIdx.x;
    float m = -3.0e38f;
    for (int p = 0; p < PNUM; ++p)
        m = fmaxf(m, h1[((size_t)(b * PNUM + p)) * 512 + d]);
    pooled[b * 512 + d] = m;
}

template <int ELU>
__global__ void head_fc_kernel(const float* __restrict__ in, const float* __restrict__ W,
                               const float* __restrict__ bias, float* __restrict__ out) {
    int t = blockIdx.x * 256 + threadIdx.x;  // 0..4095
    int r = t >> 9, n = t & 511;
    float acc = bias[n];
    for (int d = 0; d < 512; ++d) acc += in[r * 512 + d] * W[d * 512 + n];
    if (ELU) acc = acc > 0.0f ? acc : (__expf(acc) - 1.0f);
    out[t] = acc;
}

// ---------------- launch ----------------
extern "C" void kernel_launch(void* const* d_in, const int* in_sizes, int n_in,
                              void* d_out, int out_size, void* d_ws, size_t ws_size,
                              hipStream_t stream) {
    const float* pos    = (const float*)d_in[0];
    const float* W_ffm  = (const float*)d_in[2];
    const float* b_ffm  = (const float*)d_in[3];
    const float* W_lin  = (const float*)d_in[4];
    const float* W_src  = (const float*)d_in[5];
    const float* W_dst  = (const float*)d_in[6];
    const float* W_pos  = (const float*)d_in[7];
    const float* b_pos  = (const float*)d_in[8];
    const float* W_attn = (const float*)d_in[9];
    const float* b_attn = (const float*)d_in[10];
    const float* W_out  = (const float*)d_in[11];
    const float* b_out  = (const float*)d_in[12];
    const float* resc   = (const float*)d_in[13];
    const float* W_r1   = (const float*)d_in[14];
    const float* b_r1   = (const float*)d_in[15];
    const float* W_r2   = (const float*)d_in[16];
    const float* b_r2   = (const float*)d_in[17];
    const float* W_r3   = (const float*)d_in[18];
    const float* b_r3   = (const float*)d_in[19];
    const float* W_r4   = (const float*)d_in[20];
    const float* b_r4   = (const float*)d_in[21];
    float* outp = (float*)d_out;

    // bump allocator on workspace
    size_t off = 0;
    auto alloc = [&](size_t bytes) -> void* {
        void* p = (char*)d_ws + off;
        off += (bytes + 255) & ~(size_t)255;
        return p;
    };
    int*            nbr    = (int*)           alloc((size_t)NTOT * KNN * 4);
    float*          x_f    = (float*)         alloc((size_t)NTOT * DIM * 4);
    unsigned short* x_bf   = (unsigned short*)alloc((size_t)NTOT * DIM * 2);
    float*          xl     = (float*)         alloc((size_t)NTOT * DIM * 4);
    float*          u_f    = (float*)         alloc((size_t)NTOT * DIM * 4);
    float*          v_f    = (float*)         alloc((size_t)NTOT * DIM * 4);
    unsigned short* g_bf   = (unsigned short*)alloc((size_t)NTOT * DIM * 2);
    float*          h1     = (float*)         alloc((size_t)NTOT * DIM * 4);
    unsigned short* WT_lin = (unsigned short*)alloc((size_t)DIM * DIM * 2);
    unsigned short* WT_att = (unsigned short*)alloc((size_t)DIM * DIM * 2);
    unsigned short* WT_out = (unsigned short*)alloc((size_t)DIM * DIM * 2);
    unsigned short* Wsrc_b = (unsigned short*)alloc((size_t)DIM * DIM * 2);
    unsigned short* Wdst_b = (unsigned short*)alloc((size_t)DIM * DIM * 2);
    unsigned short* WsaT   = (unsigned short*)alloc((size_t)DIM * DIM * 2);
    unsigned short* WdaT   = (unsigned short*)alloc((size_t)DIM * DIM * 2);
    float*          Wpa    = (float*)         alloc(3 * DIM * 4);
    float*          bpa    = (float*)         alloc(DIM * 4);
    float*          pooled = (float*)         alloc(BNUM * DIM * 4);
    float*          h2     = (float*)         alloc(BNUM * DIM * 4);
    float*          h3     = (float*)         alloc(BNUM * DIM * 4);

    const int SQ = DIM * DIM;               // 262144
    const int SQB = SQ / 256;               // 1024 blocks for 512x512 elementwise

    // graph build + input embedding
    knn_kernel<<<BNUM, 512, 0, stream>>>(pos, nbr);
    ffm_kernel<<<(NTOT * DIM) / 256, 256, 0, stream>>>(pos, W_ffm, b_ffm, x_f, x_bf);

    for (int l = 0; l < LAY; ++l) {
        const float* Wl_lin  = W_lin  + (size_t)l * SQ;
        const float* Wl_src  = W_src  + (size_t)l * SQ;
        const float* Wl_dst  = W_dst  + (size_t)l * SQ;
        const float* Wl_pos  = W_pos  + (size_t)l * 3 * DIM;
        const float* bl_pos  = b_pos  + (size_t)l * DIM;
        const float* Wl_attn = W_attn + (size_t)l * SQ;
        const float* bl_attn = b_attn + (size_t)l * DIM;
        const float* Wl_out  = W_out  + (size_t)l * SQ;
        const float* bl_out  = b_out  + (size_t)l * DIM;

        // weight prep (bf16, transposed where used as B-operand)
        transpose_bf16_kernel<<<SQB, 256, 0, stream>>>(Wl_lin,  WT_lin);
        transpose_bf16_kernel<<<SQB, 256, 0, stream>>>(Wl_attn, WT_att);
        transpose_bf16_kernel<<<SQB, 256, 0, stream>>>(Wl_out,  WT_out);
        convert_bf16_kernel<<<SQB, 256, 0, stream>>>(Wl_src, Wsrc_b, SQ);
        convert_bf16_kernel<<<SQB, 256, 0, stream>>>(Wl_dst, Wdst_b, SQ);

        // fused weights: WsaT = (W_src @ W_attn)^T, WdaT = (W_dst @ W_attn)^T
        gemm_wmma<2><<<dim3(DIM / 128, DIM / 64), 256, 0, stream>>>(
            Wsrc_b, WT_att, nullptr, nullptr, WsaT, nullptr, nullptr, nullptr, DIM);
        gemm_wmma<2><<<dim3(DIM / 128, DIM / 64), 256, 0, stream>>>(
            Wdst_b, WT_att, nullptr, nullptr, WdaT, nullptr, nullptr, nullptr, DIM);
        // Wpa = W_pos @ W_attn ; bpa = b_pos @ W_attn + b_attn
        wpa_kernel<<<8, 256, 0, stream>>>(Wl_pos, bl_pos, Wl_attn, bl_attn, Wpa, bpa);

        // big GEMMs: xl = x@W_lin ; u = x@(W_dst@W_attn) ; v = x@(W_src@W_attn)
        dim3 gg(NTOT / 128, DIM / 64);
        gemm_wmma<0><<<gg, 256, 0, stream>>>(x_bf, WT_lin, nullptr, xl, nullptr,
                                             nullptr, nullptr, nullptr, NTOT);
        gemm_wmma<0><<<gg, 256, 0, stream>>>(x_bf, WdaT, nullptr, u_f, nullptr,
                                             nullptr, nullptr, nullptr, NTOT);
        gemm_wmma<0><<<gg, 256, 0, stream>>>(x_bf, WsaT, nullptr, v_f, nullptr,
                                             nullptr, nullptr, nullptr, NTOT);

        // softmax-weighted aggregation + ELU -> g (bf16)
        attn_kernel<<<dim3(NTOT, 2), 256, 0, stream>>>(pos, nbr, u_f, v_f, xl,
                                                       Wpa, bpa, Wl_pos, bl_pos, g_bf);

        // x = x + res_coef[l] * (g @ W_out + b_out)   (also refresh x_bf)
        gemm_wmma<1><<<gg, 256, 0, stream>>>(g_bf, WT_out, bl_out, x_f, nullptr,
                                             x_f, resc + l, x_bf, NTOT);
    }

    // head: h1 = x @ W_r1 + b_r1 (WMMA), pool, then tiny FCs
    transpose_bf16_kernel<<<SQB, 256, 0, stream>>>(W_r1, WT_lin);
    gemm_wmma<0><<<dim3(NTOT / 128, DIM / 64), 256, 0, stream>>>(
        x_bf, WT_lin, b_r1, h1, nullptr, nullptr, nullptr, nullptr, NTOT);
    pool_max_kernel<<<dim3(BNUM, 2), 256, 0, stream>>>(h1, pooled);
    head_fc_kernel<1><<<16, 256, 0, stream>>>(pooled, W_r2, b_r2, h2);
    head_fc_kernel<1><<<16, 256, 0, stream>>>(h2, W_r3, b_r3, h3);
    head_fc_kernel<0><<<16, 256, 0, stream>>>(h3, W_r4, b_r4, outp);
}
